// GCNEncoder_18176301596816
// MI455X (gfx1250) — matrix operand order repaired and checked by reading, not compiled, not run on hardware
//
#include <hip/hip_runtime.h>

#define DF  128
#define BR  128   // rows per GEMM block (amortizes W staging 8x vs 16-row blocks)
#define SAP 136   // LDS pitch for A tile (halves), padded to dodge bank conflicts
#define SWP 136   // LDS pitch for Wt (halves)

typedef __attribute__((ext_vector_type(16))) __bf16 v16bf;
typedef __attribute__((ext_vector_type(8)))  __bf16 v8bf;
typedef __attribute__((ext_vector_type(8)))  float  v8f;

__device__ __forceinline__ __bf16 f2bf(float f) {
    unsigned u = __builtin_bit_cast(unsigned, f);
    unsigned r = (u + 0x7FFFu + ((u >> 16) & 1u)) >> 16;   // round-to-nearest-even
    unsigned short s = (unsigned short)r;
    return __builtin_bit_cast(__bf16, s);
}

// ---------------------------------------------------------------------------
// Dense projection: H = X @ W   (fp32 in/out, bf16 WMMA, fp32 accum)
// One block = 128 rows x 128 cols. 8 waves; each wave owns a 16-col strip,
// preloads its B fragments once, then sweeps 8 row sub-tiles (4 WMMAs each).
// ---------------------------------------------------------------------------
__global__ __launch_bounds__(256) void gcn_gemm_wmma(
    const float* __restrict__ X, const float* __restrict__ W,
    float* __restrict__ H, int nrows)
{
    __shared__ __bf16 sA[BR * SAP];     // ~34.8 KB
    __shared__ __bf16 sW[128 * SWP];    // ~34.8 KB

    const int tid  = threadIdx.x;
    const int row0 = blockIdx.x * BR;

    // Stage W transposed: sW[n*SWP + k] = bf16(W[k*128 + n])  (B needs K-contiguous)
    for (int idx = tid; idx < 128 * 128; idx += 256) {
        int k = idx >> 7, n = idx & 127;
        sW[n * SWP + k] = f2bf(W[idx]);
    }
    // Stage A tile: BR rows of X (zero-fill past end)
    for (int idx = tid; idx < BR * 128; idx += 256) {
        int r = idx >> 7, c = idx & 127;
        int gr = row0 + r;
        float v = (gr < nrows) ? X[(size_t)gr * DF + c] : 0.0f;
        sA[r * SAP + c] = f2bf(v);
    }
    __syncthreads();

    const int wave  = tid >> 5;
    const int lane  = tid & 31;
    const int nTile = wave * 16;
    const int m     = lane & 15;
    const int hi    = lane >> 4;       // 0: low-K half, 1: high-K half
    const int n     = nTile + m;       // output column owned by this lane
    const int mb    = hi * 8;          // row base within sub-tile for C/D

    // Preload B fragments for all 4 K-steps (reused by every row sub-tile).
    // 32x16 bf16 B: lane<16 -> col n, K kt..kt+15; lane>=16 -> K kt+16..kt+31.
    v16bf bfrag[4];
    #pragma unroll
    for (int kk = 0; kk < 4; ++kk) {
        const __bf16* bp = &sW[n * SWP + kk * 32 + hi * 16];
        v8bf b0 = *(const v8bf*)bp;
        v8bf b1 = *(const v8bf*)(bp + 8);
        #pragma unroll
        for (int i = 0; i < 8; ++i) { bfrag[kk][i] = b0[i]; bfrag[kk][i + 8] = b1[i]; }
    }

    for (int rt = 0; rt < BR / 16; ++rt) {
        const __bf16* arow = &sA[(rt * 16 + m) * SAP];
        v8f c = {};
        #pragma unroll
        for (int kk = 0; kk < 4; ++kk) {
            // 16x32 bf16 A: lane<16 holds K kt..kt+15; lane>=16 holds K kt+16..kt+31
            const __bf16* ap = arow + kk * 32 + hi * 16;
            v8bf a0 = *(const v8bf*)ap;
            v8bf a1 = *(const v8bf*)(ap + 8);
            v16bf a;
            #pragma unroll
            for (int i = 0; i < 8; ++i) { a[i] = a0[i]; a[i + 8] = a1[i]; }
            c = __builtin_amdgcn_wmma_f32_16x16x32_bf16(
                    /*neg_a=*/false, a, /*neg_b=*/false, bfrag[kk],
                    /*c_mod=*/(short)0, c, /*reuse_a=*/false, /*reuse_b=*/false);
        }

        // C/D layout: lane<16 -> N=lane, M=j; lane>=16 -> N=lane-16, M=8+j
        const int gr0 = row0 + rt * 16 + mb;
        if (gr0 + 8 <= nrows) {               // fast path: straight-line stores
            float* hp = H + (size_t)gr0 * DF + n;
            #pragma unroll
            for (int j = 0; j < 8; ++j) hp[(size_t)j * DF] = c[j];
        } else {                              // tail block only
            #pragma unroll
            for (int j = 0; j < 8; ++j) {
                int gr = gr0 + j;
                if (gr < nrows) H[(size_t)gr * DF + n] = c[j];
            }
        }
    }
}

// ---------------------------------------------------------------------------
// Graph plumbing
// ---------------------------------------------------------------------------
__global__ void init_deg(float* __restrict__ deg, int n) {
    int i = blockIdx.x * blockDim.x + threadIdx.x;
    if (i < n) deg[i] = 1.0f;                      // self-loop pre-counted
}

__global__ void accum_deg(const int* __restrict__ dst, float* __restrict__ deg, int e) {
    int i = blockIdx.x * blockDim.x + threadIdx.x;
    if (i < e) atomicAdd(&deg[dst[i]], 1.0f);
}

__global__ void calc_dis(const float* __restrict__ deg, float* __restrict__ dis, int n) {
    int i = blockIdx.x * blockDim.x + threadIdx.x;
    if (i < n) dis[i] = rsqrtf(deg[i]);            // deg >= 1 always (self-loops)
}

__global__ void calc_norm(const int* __restrict__ src, const int* __restrict__ dst,
                          const float* __restrict__ dis, float* __restrict__ nrm, int e) {
    int i = blockIdx.x * blockDim.x + threadIdx.x;
    if (i < e) nrm[i] = dis[src[i]] * dis[dst[i]];
}

__global__ void zero_f32(float4* __restrict__ buf, long n4) {
    long i = (long)blockIdx.x * blockDim.x + threadIdx.x;
    if (i < n4) buf[i] = make_float4(0.f, 0.f, 0.f, 0.f);
}

// One wave per edge: 32 lanes x float4 = 128 features. h/agg are L2-resident (51MB each).
__global__ __launch_bounds__(256) void gcn_scatter(
    const float* __restrict__ h, const int* __restrict__ src,
    const int* __restrict__ dst, const float* __restrict__ nrm,
    float* __restrict__ agg, int nedges)
{
    const int lane = threadIdx.x & 31;
    const int wid  = blockIdx.x * (blockDim.x >> 5) + (threadIdx.x >> 5);
    const int nw   = gridDim.x * (blockDim.x >> 5);
    for (int e = wid; e < nedges; e += nw) {
        const int   s = src[e];
        const int   d = dst[e];
        const float w = nrm[e];
        const float4 hv = *(const float4*)(h + (size_t)s * DF + lane * 4);
        float* ap = agg + (size_t)d * DF + lane * 4;
        atomicAdd(ap + 0, hv.x * w);
        atomicAdd(ap + 1, hv.y * w);
        atomicAdd(ap + 2, hv.z * w);
        atomicAdd(ap + 3, hv.w * w);
    }
}

// out = [relu](agg + h*dis^2 + b)   -- fuses self-loop term + bias (+ReLU)
__global__ void gcn_finish(const float* __restrict__ agg, const float* __restrict__ h,
                           const float* __restrict__ dis, const float* __restrict__ bias,
                           float* __restrict__ out, long total, int relu)
{
    long idx = (long)blockIdx.x * blockDim.x + threadIdx.x;
    if (idx < total) {
        int i = (int)(idx >> 7);
        int f = (int)(idx & 127);
        float di = dis[i];
        float v = agg[idx] + h[idx] * di * di + bias[f];
        out[idx] = relu ? fmaxf(v, 0.0f) : v;
    }
}

// ---------------------------------------------------------------------------
extern "C" void kernel_launch(void* const* d_in, const int* in_sizes, int n_in,
                              void* d_out, int out_size, void* d_ws, size_t ws_size,
                              hipStream_t stream)
{
    const float* x  = (const float*)d_in[0];
    const int*   ei = (const int*)  d_in[1];
    const float* W1 = (const float*)d_in[2];
    const float* b1 = (const float*)d_in[3];
    const float* W2 = (const float*)d_in[4];
    const float* b2 = (const float*)d_in[5];
    const float* W3 = (const float*)d_in[6];
    const float* b3 = (const float*)d_in[7];

    const int N = in_sizes[0] / DF;
    const int E = in_sizes[1] / 2;
    const int* src = ei;
    const int* dst = ei + E;

    // Workspace carve (~110 MB): deg, dis, norm, agg, act. h lives in d_out.
    char* p = (char*)d_ws;
    auto carve = [&p](size_t bytes) {
        void* r = (void*)p;
        p += (bytes + 255) & ~(size_t)255;
        return r;
    };
    float* deg = (float*)carve((size_t)N * 4);
    float* dis = (float*)carve((size_t)N * 4);
    float* nrm = (float*)carve((size_t)E * 4);
    float* agg = (float*)carve((size_t)N * DF * 4);
    float* act = (float*)carve((size_t)N * DF * 4);
    float* h   = (float*)d_out;                 // reuse output as GEMM scratch

    const long totalF  = (long)N * DF;
    const long total4  = totalF / 4;
    const int  TPB     = 256;
    const int  gN      = (N + TPB - 1) / TPB;
    const int  gE      = (E + TPB - 1) / TPB;
    const int  gF      = (int)((totalF + TPB - 1) / TPB);
    const int  g4      = (int)((total4 + TPB - 1) / TPB);
    const int  gGemm   = (N + BR - 1) / BR;
    const int  gScat   = 4096;                  // 32768 waves, grid-stride over edges

    // --- normalization (computed once, reused by all layers) ---
    init_deg <<<gN, TPB, 0, stream>>>(deg, N);
    accum_deg<<<gE, TPB, 0, stream>>>(dst, deg, E);
    calc_dis <<<gN, TPB, 0, stream>>>(deg, dis, N);
    calc_norm<<<gE, TPB, 0, stream>>>(src, dst, dis, nrm, E);

    // --- layer 1 ---
    gcn_gemm_wmma<<<gGemm, TPB, 0, stream>>>(x, W1, h, N);
    zero_f32     <<<g4,    TPB, 0, stream>>>((float4*)agg, total4);
    gcn_scatter  <<<gScat, TPB, 0, stream>>>(h, src, dst, nrm, agg, E);
    gcn_finish   <<<gF,    TPB, 0, stream>>>(agg, h, dis, b1, act, totalF, 1);

    // --- layer 2 ---
    gcn_gemm_wmma<<<gGemm, TPB, 0, stream>>>(act, W2, h, N);
    zero_f32     <<<g4,    TPB, 0, stream>>>((float4*)agg, total4);
    gcn_scatter  <<<gScat, TPB, 0, stream>>>(h, src, dst, nrm, agg, E);
    gcn_finish   <<<gF,    TPB, 0, stream>>>(agg, h, dis, b2, act, totalF, 1);

    // --- layer 3 (no ReLU) ---
    gcn_gemm_wmma<<<gGemm, TPB, 0, stream>>>(act, W3, h, N);
    zero_f32     <<<g4,    TPB, 0, stream>>>((float4*)agg, total4);
    gcn_scatter  <<<gScat, TPB, 0, stream>>>(h, src, dst, nrm, agg, E);
    gcn_finish   <<<gF,    TPB, 0, stream>>>(agg, h, dis, b3, (float*)d_out, totalF, 0);
}
